// Block_23106924052766
// MI455X (gfx1250) — matrix operand order repaired and checked by reading, not compiled, not run on hardware
//
#include <hip/hip_runtime.h>
#include <math.h>

// ---------------- types ----------------
typedef __bf16 bf16;
typedef __attribute__((ext_vector_type(16))) __bf16 v16bf;
typedef __attribute__((ext_vector_type(8)))  float  v8f;

#define DEVI __device__ __forceinline__

// ---------------- problem constants ----------------
#define Hh    56
#define Cc    128
#define HEADS 4
#define WSz   7
#define SHIFT 3
#define Ntok  49          // tokens per window
#define TOK   (Hh*Hh)     // 3136 tokens per image
#define HID   512
#define QK_SCALE 0.17677669529663687f  // 32^-0.5

// ws layout (bf16, transposed weights W^T[n][k])
#define WS_QKVT 0         // [384][128]
#define WS_PROJT 49152    // [128][128]
#define WS_FC1T 65536     // [512][128]
#define WS_FC2T 131072    // [128][512]
#define WS_ELEMS 196608   // bf16 elements total

// =====================================================================
// Fragment loaders (wave32). A/B use this code's contiguous k order (same in both); C/D is the ISA layout.
// A 16x32 bf16: lane = half*16+m ; elem j -> K = half*16 + j
//   (one contiguous 32B chunk per lane when source is row-major)
// B 32x16 bf16: lane = khalf*16+n ; elem j -> K = khalf*16 + j
// C/D 16x16 f32: lane = half*16+n ; vgpr r -> M = r + 8*half
// =====================================================================
DEVI v16bf load_a(const bf16* m, int row0, int k0, int ld) {
  int lane = threadIdx.x & 31;
  int half = lane >> 4, r = lane & 15;
  const bf16* base = m + (size_t)(row0 + r) * ld + k0 + half * 16;
  v16bf a;
#pragma unroll
  for (int j = 0; j < 16; ++j) {
    a[j] = base[j];
  }
  return a;
}

// B[k][n] = m[(n0+n)*ld + k0+k]  (K-contiguous source: transposed weights,
// row-major keys for q@k^T, Vt for P@V) -> 32B contiguous per lane
DEVI v16bf load_b_t(const bf16* m, int k0, int n0, int ld) {
  int lane = threadIdx.x & 31;
  int khalf = lane >> 4, n = lane & 15;
  const bf16* base = m + (size_t)(n0 + n) * ld + k0 + khalf * 16;
  v16bf b;
#pragma unroll
  for (int j = 0; j < 16; ++j) b[j] = base[j];
  return b;
}

// Fallback: B[k][n] = (bf16) m[(k0+k)*ld + n0+n], fp32 row-major source
DEVI v16bf load_b_f32(const float* m, int k0, int n0, int ld) {
  int lane = threadIdx.x & 31;
  int khalf = lane >> 4, n = lane & 15;
  const float* base = m + (size_t)(k0 + khalf * 16) * ld + n0 + n;
  v16bf b;
#pragma unroll
  for (int j = 0; j < 16; ++j) b[j] = (bf16)base[(size_t)j * ld];
  return b;
}

DEVI v8f wmma_bf16(v16bf a, v16bf b, v8f c) {
  return __builtin_amdgcn_wmma_f32_16x16x32_bf16(false, a, false, b, (short)0, c,
                                                 false, false);
}

DEVI int region9(int gr, int gc) {  // shifted-window mask region id
  int rr = (gr < Hh - WSz) ? 0 : (gr < Hh - SHIFT ? 1 : 2);
  int rc = (gc < Hh - WSz) ? 0 : (gc < Hh - SHIFT ? 1 : 2);
  return rr * 3 + rc;
}

// =====================================================================
// Prep: convert weights to bf16, transposed (W^T[n][k]) into d_ws.
// =====================================================================
__global__ void __launch_bounds__(256)
prep_weights(const float* __restrict__ qkv_w, const float* __restrict__ proj_w,
             const float* __restrict__ fc1_w, const float* __restrict__ fc2_w,
             bf16* __restrict__ ws) {
  int i = blockIdx.x * 256 + threadIdx.x;
  if (i < 49152) {                       // qkvT [384][128] <- qkv_w [128][384]
    int n = i >> 7, k = i & 127;
    ws[WS_QKVT + i] = (bf16)qkv_w[k * 384 + n];
  } else if (i < 65536) {                // projT [128][128] <- proj_w [128][128]
    int j = i - 49152;
    int n = j >> 7, k = j & 127;
    ws[WS_PROJT + j] = (bf16)proj_w[k * 128 + n];
  } else if (i < 131072) {               // fc1T [512][128] <- fc1_w [128][512]
    int j = i - 65536;
    int n = j >> 7, k = j & 127;
    ws[WS_FC1T + j] = (bf16)fc1_w[k * 512 + n];
  } else if (i < WS_ELEMS) {             // fc2T [128][512] <- fc2_w [512][128]
    int j = i - 131072;
    int n = j >> 9, k = j & 511;
    ws[WS_FC2T + j] = (bf16)fc2_w[k * 128 + n];
  }
}

// =====================================================================
// Kernel 1: one block per (image, window).  128 threads = 4 waves.
// LDS: Ahw/O bf16 64x128 @0 (16K) | QK bf16 64x256 @16K (32K)
//      Vt bf16 128x64 @48K (16K)  | S f32 4x64x64 @64K (64K, overlaps xstage)
// =====================================================================
template <bool USE_WS>
__global__ void __launch_bounds__(128)
swin_attn_kernel(const float* __restrict__ x,
                 const float* __restrict__ n1g, const float* __restrict__ n1b,
                 const float* __restrict__ qkv_w, const float* __restrict__ qkv_b,
                 const float* __restrict__ proj_w, const float* __restrict__ proj_b,
                 const float* __restrict__ rpb,
                 const bf16* __restrict__ qkvT, const bf16* __restrict__ projT,
                 float* __restrict__ xout) {
  extern __shared__ char smem[];
  bf16*  Ahw    = (bf16*)smem;                 // [64][128] LN'd tokens, later O
  bf16*  QK     = (bf16*)(smem + 16 * 1024);   // [64][256] q(scaled) | k
  bf16*  Vt     = (bf16*)(smem + 48 * 1024);   // [128][64]  v transposed
  float* Sbuf   = (float*)(smem + 64 * 1024);  // [4][64][64] scores (P overlays)
  float* xstage = (float*)(smem + 64 * 1024);  // [64][128] raw tokens (early)

  const int tid = threadIdx.x, lane = tid & 31, wid = tid >> 5;
  const int blk = blockIdx.x;
  const int b = blk >> 6, w = blk & 63;
  const int wi = w >> 3, wj = w & 7;
  const float* xb = x + (size_t)b * TOK * Cc;

  // ---- stage tokens (cyclic -SHIFT roll + window partition) ----
  for (int t = wid; t < 64; t += 4) {
    if (t < Ntok) {
      int sr = (wi * WSz + t / WSz + SHIFT) % Hh;
      int sc = (wj * WSz + t % WSz + SHIFT) % Hh;
      ((float4*)(xstage + t * Cc))[lane] =
          ((const float4*)(xb + (size_t)(sr * Hh + sc) * Cc))[lane];
    } else {
      float4 z = {0.f, 0.f, 0.f, 0.f};
      ((float4*)(xstage + t * Cc))[lane] = z;
    }
  }
  __syncthreads();

  // ---- LayerNorm1 -> bf16 A ----
  if (tid < 64) {
    int t = tid;
    if (t < Ntok) {
      const float* row = xstage + t * Cc;
      float m = 0.f;
      for (int c = 0; c < Cc; ++c) m += row[c];
      m *= (1.f / Cc);
      float v = 0.f;
      for (int c = 0; c < Cc; ++c) { float d = row[c] - m; v += d * d; }
      v *= (1.f / Cc);
      float inv = rsqrtf(v + 1e-5f);
      for (int c = 0; c < Cc; ++c)
        Ahw[t * Cc + c] = (bf16)((row[c] - m) * inv * n1g[c] + n1b[c]);
    } else {
      for (int c = 0; c < Cc; ++c) Ahw[t * Cc + c] = (bf16)0.f;
    }
  }
  __syncthreads();

  // ---- QKV GEMM: (64x128) @ (128x384); wave = M-tile ----
  {
    v16bf afr[4];
#pragma unroll
    for (int k = 0; k < 4; ++k) afr[k] = load_a(Ahw, wid * 16, k * 32, Cc);
    for (int nt = 0; nt < 24; ++nt) {
      v8f acc = {};
#pragma unroll
      for (int k = 0; k < 4; ++k) {
        v16bf bb = USE_WS ? load_b_t(qkvT, k * 32, nt * 16, Cc)
                          : load_b_f32(qkv_w, k * 32, nt * 16, 384);
        acc = wmma_bf16(afr[k], bb, acc);
      }
      int n = lane & 15, half = lane >> 4;
      int col = nt * 16 + n;
      float bias = qkv_b[col];
      if (nt < 16) {  // q (pre-scaled) and k -> row-major QK
        float scl = (col < Cc) ? QK_SCALE : 1.0f;
#pragma unroll
        for (int r = 0; r < 8; ++r) {
          int row = wid * 16 + r + 8 * half;
          QK[row * 256 + col] = (bf16)((acc[r] + bias) * scl);
        }
      } else {        // v -> transposed Vt[d][token]; 8 contiguous rows per lane
        bf16* vcol = Vt + (size_t)(col - 256) * 64 + wid * 16 + 8 * half;
#pragma unroll
        for (int r = 0; r < 8; ++r) vcol[r] = (bf16)(acc[r] + bias);
      }
    }
  }
  __syncthreads();

  // ---- scores S = q @ k^T + rel-pos bias + shift mask ; wave = head ----
  {
    int h = wid;
    v16bf bk[4];
#pragma unroll
    for (int nt = 0; nt < 4; ++nt)
      bk[nt] = load_b_t(QK, Cc + h * 32, nt * 16, 256);  // B[d][key] = k[key][d]
    for (int mt = 0; mt < 4; ++mt) {
      v16bf aq = load_a(QK, mt * 16, h * 32, 256);       // scaled q rows
#pragma unroll
      for (int nt = 0; nt < 4; ++nt) {
        v8f acc = {};
        acc = wmma_bf16(aq, bk[nt], acc);
        int n = lane & 15, half = lane >> 4;
        int kidx = nt * 16 + n;
#pragma unroll
        for (int r = 0; r < 8; ++r) {
          int qidx = mt * 16 + r + 8 * half;
          float val = acc[r];
          if (qidx < Ntok && kidx < Ntok) {
            int qr = qidx / WSz, qc = qidx % WSz;
            int kr = kidx / WSz, kc = kidx % WSz;
            int rel = (qr - kr + WSz - 1) * (2 * WSz - 1) + (qc - kc + WSz - 1);
            val += rpb[rel * HEADS + h];
            if (region9(wi * WSz + qr, wj * WSz + qc) !=
                region9(wi * WSz + kr, wj * WSz + kc))
              val += -100.0f;
          } else {
            val = -1e9f;
          }
          Sbuf[(h * 64 + qidx) * 64 + kidx] = val;
        }
      }
    }
  }
  __syncthreads();

  // ---- softmax over keys (<49), convert to bf16 P in place ----
  for (int i = tid; i < HEADS * 64; i += 128) {
    int row = i & 63;
    float* srow = Sbuf + (size_t)i * 64;
    bf16*  prow = (bf16*)srow;  // in-place: write 2B behind the 4B reads
    if (row < Ntok) {
      float mx = -1e30f;
      for (int m = 0; m < Ntok; ++m) mx = fmaxf(mx, srow[m]);
      float sum = 0.f;
      for (int m = 0; m < Ntok; ++m) sum += expf(srow[m] - mx);
      float inv = 1.f / sum;
      for (int m = 0; m < 64; ++m) {
        float p = (m < Ntok) ? expf(srow[m] - mx) * inv : 0.f;
        prow[m] = (bf16)p;
      }
    } else {
      for (int m = 0; m < 64; ++m) prow[m] = (bf16)0.f;
    }
  }
  __syncthreads();

  // ---- O = P @ v ; wave = head ; O reuses Ahw region ----
  {
    int h = wid;
    const bf16* P = (const bf16*)(Sbuf + (size_t)h * 64 * 64);  // ld = 128 bf16
    v16bf bv[4];
#pragma unroll
    for (int kt = 0; kt < 2; ++kt)
#pragma unroll
      for (int nt = 0; nt < 2; ++nt)   // B[token][d] = Vt[d][token] contiguous
        bv[kt * 2 + nt] = load_b_t(Vt, kt * 32, h * 32 + nt * 16, 64);
    for (int mt = 0; mt < 4; ++mt) {
#pragma unroll
      for (int nt = 0; nt < 2; ++nt) {
        v8f acc = {};
#pragma unroll
        for (int kt = 0; kt < 2; ++kt)
          acc = wmma_bf16(load_a(P, mt * 16, kt * 32, 128), bv[kt * 2 + nt], acc);
        int n = lane & 15, half = lane >> 4;
        int col = h * 32 + nt * 16 + n;
#pragma unroll
        for (int r = 0; r < 8; ++r)
          Ahw[(mt * 16 + r + 8 * half) * Cc + col] = (bf16)acc[r];
      }
    }
  }
  __syncthreads();

  // ---- proj + reverse shift merge + residual ----
  for (int mt = 0; mt < 4; ++mt) {
    v16bf ao[4];
#pragma unroll
    for (int k = 0; k < 4; ++k) ao[k] = load_a(Ahw, mt * 16, k * 32, Cc);
#pragma unroll
    for (int nti = 0; nti < 2; ++nti) {
      int nt = wid * 2 + nti;
      v8f acc = {};
#pragma unroll
      for (int k = 0; k < 4; ++k) {
        v16bf bb = USE_WS ? load_b_t(projT, k * 32, nt * 16, Cc)
                          : load_b_f32(proj_w, k * 32, nt * 16, Cc);
        acc = wmma_bf16(ao[k], bb, acc);
      }
      int n = lane & 15, half = lane >> 4;
      int col = nt * 16 + n;
      float pb = proj_b[col];
#pragma unroll
      for (int r = 0; r < 8; ++r) {
        int row = mt * 16 + r + 8 * half;
        if (row < Ntok) {
          int sr = (wi * WSz + row / WSz + SHIFT) % Hh;
          int sc = (wj * WSz + row % WSz + SHIFT) % Hh;
          size_t idx = (size_t)b * TOK * Cc + (size_t)(sr * Hh + sc) * Cc + col;
          xout[idx] = x[idx] + acc[r] + pb;
        }
      }
    }
  }
}

// =====================================================================
// Kernel 2: MLP block, 64 tokens per block.
// LDS: xstage f32 64x128 @0 (32K) | A bf16 @32K (16K) | H1 bf16 64x512 @48K (64K)
// =====================================================================
template <bool USE_WS>
__global__ void __launch_bounds__(128)
swin_mlp_kernel(const float* __restrict__ n2g, const float* __restrict__ n2b,
                const float* __restrict__ fc1_w, const float* __restrict__ fc1_b,
                const float* __restrict__ fc2_w, const float* __restrict__ fc2_b,
                const bf16* __restrict__ fc1T, const bf16* __restrict__ fc2T,
                float* __restrict__ xio) {
  extern __shared__ char smem[];
  float* xstage = (float*)smem;                // [64][128]
  bf16*  A      = (bf16*)(smem + 32 * 1024);   // [64][128]
  bf16*  H1     = (bf16*)(smem + 48 * 1024);   // [64][512]

  const int tid = threadIdx.x, lane = tid & 31, wid = tid >> 5;
  float* xp = xio + (size_t)blockIdx.x * 64 * Cc;

  for (int i = tid; i < 64 * Cc / 4; i += 128)
    ((float4*)xstage)[i] = ((const float4*)xp)[i];
  __syncthreads();

  if (tid < 64) {
    const float* row = xstage + tid * Cc;
    float m = 0.f;
    for (int c = 0; c < Cc; ++c) m += row[c];
    m *= (1.f / Cc);
    float v = 0.f;
    for (int c = 0; c < Cc; ++c) { float d = row[c] - m; v += d * d; }
    v *= (1.f / Cc);
    float inv = rsqrtf(v + 1e-5f);
    for (int c = 0; c < Cc; ++c)
      A[tid * Cc + c] = (bf16)((row[c] - m) * inv * n2g[c] + n2b[c]);
  }
  __syncthreads();

  // ---- fc1 + exact GELU ; wave = M-tile ----
  {
    v16bf afr[4];
#pragma unroll
    for (int k = 0; k < 4; ++k) afr[k] = load_a(A, wid * 16, k * 32, Cc);
    for (int nt = 0; nt < 32; ++nt) {
      v8f acc = {};
#pragma unroll
      for (int k = 0; k < 4; ++k) {
        v16bf bb = USE_WS ? load_b_t(fc1T, k * 32, nt * 16, Cc)
                          : load_b_f32(fc1_w, k * 32, nt * 16, HID);
        acc = wmma_bf16(afr[k], bb, acc);
      }
      int n = lane & 15, half = lane >> 4;
      int col = nt * 16 + n;
      float bb = fc1_b[col];
#pragma unroll
      for (int r = 0; r < 8; ++r) {
        float v = acc[r] + bb;
        v = 0.5f * v * (1.f + erff(v * 0.70710678118654752f));
        H1[(wid * 16 + r + 8 * half) * HID + col] = (bf16)v;
      }
    }
  }
  __syncthreads();

  // ---- fc2 + residual ; 8 live accumulators, A loaded once per K-step ----
  {
    v8f acc[8];
#pragma unroll
    for (int nt = 0; nt < 8; ++nt) acc[nt] = (v8f){};
    for (int k = 0; k < 16; ++k) {
      v16bf a2 = load_a(H1, wid * 16, k * 32, HID);
#pragma unroll
      for (int nt = 0; nt < 8; ++nt) {
        v16bf bb = USE_WS ? load_b_t(fc2T, k * 32, nt * 16, HID)
                          : load_b_f32(fc2_w, k * 32, nt * 16, Cc);
        acc[nt] = wmma_bf16(a2, bb, acc[nt]);
      }
    }
    int n = lane & 15, half = lane >> 4;
#pragma unroll
    for (int nt = 0; nt < 8; ++nt) {
      int col = nt * 16 + n;
      float bb = fc2_b[col];
#pragma unroll
      for (int r = 0; r < 8; ++r) {
        int row = wid * 16 + r + 8 * half;
        xp[row * Cc + col] = xstage[row * Cc + col] + acc[nt][r] + bb;
      }
    }
  }
}

// =====================================================================
extern "C" void kernel_launch(void* const* d_in, const int* in_sizes, int n_in,
                              void* d_out, int out_size, void* d_ws, size_t ws_size,
                              hipStream_t stream) {
  (void)in_sizes; (void)n_in; (void)out_size;
  const float* x      = (const float*)d_in[0];
  const float* n1g    = (const float*)d_in[1];
  const float* n1b    = (const float*)d_in[2];
  const float* qkv_w  = (const float*)d_in[3];
  const float* qkv_b  = (const float*)d_in[4];
  const float* proj_w = (const float*)d_in[5];
  const float* proj_b = (const float*)d_in[6];
  const float* rpb    = (const float*)d_in[7];
  const float* n2g    = (const float*)d_in[8];
  const float* n2b    = (const float*)d_in[9];
  const float* fc1_w  = (const float*)d_in[10];
  const float* fc1_b  = (const float*)d_in[11];
  const float* fc2_w  = (const float*)d_in[12];
  const float* fc2_b  = (const float*)d_in[13];
  float* xout = (float*)d_out;
  bf16*  ws   = (bf16*)d_ws;

  const bool use_ws = ws_size >= (size_t)WS_ELEMS * sizeof(bf16);

  if (use_ws) {
    (void)hipFuncSetAttribute((const void*)swin_attn_kernel<true>,
                              hipFuncAttributeMaxDynamicSharedMemorySize, 128 * 1024);
    (void)hipFuncSetAttribute((const void*)swin_mlp_kernel<true>,
                              hipFuncAttributeMaxDynamicSharedMemorySize, 112 * 1024);
    prep_weights<<<(WS_ELEMS + 255) / 256, 256, 0, stream>>>(qkv_w, proj_w, fc1_w,
                                                             fc2_w, ws);
    swin_attn_kernel<true><<<2048, 128, 128 * 1024, stream>>>(
        x, n1g, n1b, qkv_w, qkv_b, proj_w, proj_b, rpb,
        ws + WS_QKVT, ws + WS_PROJT, xout);
    swin_mlp_kernel<true><<<1568, 128, 112 * 1024, stream>>>(
        n2g, n2b, fc1_w, fc1_b, fc2_w, fc2_b, ws + WS_FC1T, ws + WS_FC2T, xout);
  } else {
    (void)hipFuncSetAttribute((const void*)swin_attn_kernel<false>,
                              hipFuncAttributeMaxDynamicSharedMemorySize, 128 * 1024);
    (void)hipFuncSetAttribute((const void*)swin_mlp_kernel<false>,
                              hipFuncAttributeMaxDynamicSharedMemorySize, 112 * 1024);
    swin_attn_kernel<false><<<2048, 128, 128 * 1024, stream>>>(
        x, n1g, n1b, qkv_w, qkv_b, proj_w, proj_b, rpb, nullptr, nullptr, xout);
    swin_mlp_kernel<false><<<1568, 128, 112 * 1024, stream>>>(
        n2g, n2b, fc1_w, fc1_b, fc2_w, fc2_b, nullptr, nullptr, xout);
  }
}
